// SMoE_22016002359372
// MI455X (gfx1250) — matrix operand order repaired and checked by reading, not compiled, not run on hardware
//
#include <hip/hip_runtime.h>
#include <cstdint>

typedef _Float16 v16h __attribute__((ext_vector_type(16)));
typedef _Float16 v8h  __attribute__((ext_vector_type(8)));
typedef float    v8f  __attribute__((ext_vector_type(8)));
typedef unsigned int u32x4 __attribute__((ext_vector_type(4)));
typedef int          i32x4 __attribute__((ext_vector_type(4)));
typedef int          i32x8 __attribute__((ext_vector_type(8)));

#define B_TOK 8192
#define D_DIM 1024
#define H_DIM 4096
#define E_EXP 8
#define EPS_COMBINE 2.220446049250313e-16f

// ---------------- workspace layout (bytes) ----------------
// w1h  : f16 [E][H][D]       @ 0          size 67108864
// w2h  : f16 [E][D][H]       @ 67108864   size 67108864
// x16  : f16 [B][D]          @ 134217728  size 16777216
// hbuf : f16 [B*2][H]        @ 150994944  size 134217728
// gateval: f32 [B*2]         @ 285212672  size 65536
// gidx : i32 [B*2]           @ 285278208  size 65536
// perm : i32 [B*2]           @ 285343744  size 65536
// counters @ 285409280: importance f32[8] | countI i32[8] | cursor i32[8] | baseI i32[8]
#define WS_W2H   67108864u
#define WS_X16   134217728u
#define WS_HBUF  150994944u
#define WS_GVAL  285212672u
#define WS_GIDX  285278208u
#define WS_PERM  285343744u
#define WS_CTR   285409280u

// -------------------------------------------------------------------
__global__ __launch_bounds__(256)
void zero_kernel(float* out, int n, int* ctr, int nctr) {
    int i = blockIdx.x * blockDim.x + threadIdx.x;
    int stride = gridDim.x * blockDim.x;
    for (int j = i; j < n; j += stride) out[j] = 0.0f;
    if (i < nctr) ctr[i] = 0;
}

__global__ __launch_bounds__(256)
void cvt_f32_to_f16_kernel(const float* __restrict__ src, _Float16* __restrict__ dst, int n) {
    int i = blockIdx.x * blockDim.x + threadIdx.x;
    int stride = gridDim.x * blockDim.x;
    for (int j = i; j < n; j += stride) dst[j] = (_Float16)src[j];
}

// -------------------------------------------------------------------
// Gating: one wave32 per token. logits = x[b] . w_gate[e], top-2, softmax.
__global__ __launch_bounds__(256)
void gate_kernel(const float* __restrict__ x, const float* __restrict__ wg,
                 float* __restrict__ gateval, int* __restrict__ gidx,
                 float* __restrict__ importance, int* __restrict__ countI) {
    const int wv = threadIdx.x >> 5;
    const int lane = threadIdx.x & 31;
    const int b = blockIdx.x * 8 + wv;
    if (b >= B_TOK) return;

    float acc[E_EXP];
#pragma unroll
    for (int e = 0; e < E_EXP; ++e) acc[e] = 0.0f;

    const float* xr = x + (size_t)b * D_DIM;
    for (int i = 0; i < D_DIM / 32; ++i) {
        float xv = xr[i * 32 + lane];
#pragma unroll
        for (int e = 0; e < E_EXP; ++e)
            acc[e] += xv * wg[e * D_DIM + i * 32 + lane];
    }
#pragma unroll
    for (int off = 16; off >= 1; off >>= 1) {
#pragma unroll
        for (int e = 0; e < E_EXP; ++e)
            acc[e] += __shfl_xor(acc[e], off, 32);
    }
    // top-2, ties -> lowest index (match lax.top_k)
    int i1 = 0; float v1 = acc[0];
#pragma unroll
    for (int e = 1; e < E_EXP; ++e) if (acc[e] > v1) { v1 = acc[e]; i1 = e; }
    int i2 = -1; float v2 = -3.4e38f;
#pragma unroll
    for (int e = 0; e < E_EXP; ++e) if (e != i1 && acc[e] > v2) { v2 = acc[e]; i2 = e; }

    float t  = __expf(v2 - v1);           // stable: v1 >= v2
    float g1 = 1.0f / (1.0f + t);
    float g2 = t / (1.0f + t);

    if (lane == 0) {
        gateval[b * 2]     = g1;  gateval[b * 2 + 1] = g2;
        gidx[b * 2]        = i1;  gidx[b * 2 + 1]    = i2;
        atomicAdd(importance + i1, g1);
        atomicAdd(importance + i2, g2);
        atomicAdd(countI + i1, 1);
        atomicAdd(countI + i2, 1);
    }
}

// -------------------------------------------------------------------
__global__ void finalize_stats_kernel(const float* __restrict__ importance,
                                      const int* __restrict__ countI,
                                      int* __restrict__ baseI, float* __restrict__ loss_out) {
    if (threadIdx.x != 0 || blockIdx.x != 0) return;
    int run = 0;
    float loadf[E_EXP];
    for (int e = 0; e < E_EXP; ++e) { baseI[e] = run; run += countI[e]; loadf[e] = (float)countI[e]; }

    float mi = 0.f, ml = 0.f;
    for (int e = 0; e < E_EXP; ++e) { mi += importance[e]; ml += loadf[e]; }
    mi *= (1.0f / E_EXP); ml *= (1.0f / E_EXP);
    float vi = 0.f, vl = 0.f;
    for (int e = 0; e < E_EXP; ++e) {
        float di = importance[e] - mi; vi += di * di;
        float dl = loadf[e] - ml;      vl += dl * dl;
    }
    vi *= (1.0f / (E_EXP - 1)); vl *= (1.0f / (E_EXP - 1));
    float cvi = vi / (mi * mi + 1e-10f);
    float cvl = vl / (ml * ml + 1e-10f);
    loss_out[0] = (cvi + cvl) * 0.01f;
}

// -------------------------------------------------------------------
__global__ __launch_bounds__(256)
void scatter_kernel(const int* __restrict__ gidx, const int* __restrict__ baseI,
                    int* __restrict__ cursor, int* __restrict__ perm) {
    int b = blockIdx.x * blockDim.x + threadIdx.x;
    if (b >= B_TOK) return;
#pragma unroll
    for (int k = 0; k < 2; ++k) {
        int e = gidx[b * 2 + k];
        int pos = atomicAdd(cursor + e, 1);
        perm[baseI[e] + pos] = b * 2 + k;   // packed (token<<1)|slot
    }
}

// -------------------------------------------------------------------
// fc1: h = gelu(x @ W1^T + b1) for one (expert, 64-token tile).
// 512 threads = 16 waves. Dynamic LDS: xtile f16[64][1024] = 131072 B.
// Token rows are gathered into LDS by the Tensor Data Mover in gather mode:
// 4 descriptors x 16 row-indices (16-bit), tracked by TENSORcnt (wave 0 only).
__global__ __launch_bounds__(512)
void moe_fc1_kernel(const _Float16* __restrict__ x16,
                    const _Float16* __restrict__ w1h,
                    const float* __restrict__ b1,
                    const int* __restrict__ perm,
                    const int* __restrict__ countI,
                    const int* __restrict__ baseI,
                    _Float16* __restrict__ hbuf) {
    extern __shared__ char smem[];
    _Float16* xtile = (_Float16*)smem;              // [64][1024]

    const int e = blockIdx.y;
    const int t = blockIdx.x;
    const int cnt = countI[e];
    if (t * 64 >= cnt) return;
    const int rows = (cnt - t * 64 < 64) ? (cnt - t * 64) : 64;
    const int base = baseI[e];

    const int tid  = threadIdx.x;
    const int lane = tid & 31;
    const int wv   = tid >> 5;
    const int lg   = lane >> 4;
    const int ln   = lane & 15;

    // ---- TDM gather: 64 token rows (f16, 2048B each) -> LDS ----
    if (wv == 0) {                        // uniform branch: only wave 0 issues TDM
        const unsigned ldsbase = (unsigned)(uintptr_t)xtile;
        const unsigned long long ga = (unsigned long long)(uintptr_t)x16;
#pragma unroll
        for (int i = 0; i < 4; ++i) {
            int pk[8];
#pragma unroll
            for (int j = 0; j < 8; ++j) {
                int r0 = i * 16 + 2 * j, r1 = r0 + 1;
                int p0 = perm[base + t * 64 + ((r0 < rows) ? r0 : 0)];
                int p1 = perm[base + t * 64 + ((r1 < rows) ? r1 : 0)];
                pk[j] = ((p0 >> 1) & 0xffff) | (((p1 >> 1) & 0xffff) << 16);
            }
            // D# group0: count=1, gather_mode (16-bit idx), lds_addr, global_addr, type=2
            u32x4 g0 = { 0x80000001u,
                         ldsbase + (unsigned)i * 32768u,
                         (unsigned)(ga & 0xffffffffu),
                         (unsigned)((ga >> 32) & 0x01ffffffu) | (2u << 30) };
            // D# group1: data_size=2B, tensor_dim0=1024, tensor_dim1=8192,
            //            tile_dim0=1024, tile_dim1=16 indices, dim0_stride=1024
            i32x8 g1 = { (int)(1u << 16),
                         (int)(1024u << 16),
                         (int)(8192u << 16),
                         (int)(1024u << 16),
                         16,
                         1024,
                         0, 0 };
            i32x4 g2 = { pk[0], pk[1], pk[2], pk[3] };
            i32x4 g3 = { pk[4], pk[5], pk[6], pk[7] };
            i32x8 gz = { 0, 0, 0, 0, 0, 0, 0, 0 };   // extra group (clang-23 6-arg form)
            __builtin_amdgcn_tensor_load_to_lds(g0, g1, g2, g3, gz, 0);
        }
        __builtin_amdgcn_s_wait_tensorcnt(0);
    }
    __syncthreads();

    const _Float16* w1e = w1h + (size_t)e * H_DIM * D_DIM;
    _Float16* hout = hbuf + (size_t)(base + t * 64) * H_DIM;

    for (int ct = 0; ct < 16; ++ct) {
        const int ncol = wv * 256 + ct * 16 + ln;
        const float bv = b1[e * H_DIM + ncol];
        v8f acc[4];
#pragma unroll
        for (int rt = 0; rt < 4; ++rt)
#pragma unroll
            for (int j = 0; j < 8; ++j) acc[rt][j] = bv;

        const _Float16* brow = w1e + (size_t)ncol * D_DIM + lg * 16;
        for (int kt = 0; kt < D_DIM / 32; ++kt) {
            const int kb = kt * 32;
            v16h Bv = *(const v16h*)(brow + kb);
#pragma unroll
            for (int rt = 0; rt < 4; ++rt) {
                const _Float16* ar = xtile + (rt * 16 + ln) * D_DIM + lg * 16 + kb;
                union { v16h v; v8h h[2]; } A;
                A.h[0] = *(const v8h*)(ar);
                A.h[1] = *(const v8h*)(ar + 8);
                acc[rt] = __builtin_amdgcn_wmma_f32_16x16x32_f16(false, A.v, false, Bv,
                                                                 (short)0, acc[rt], false, false);
            }
        }
#pragma unroll
        for (int rt = 0; rt < 4; ++rt)
#pragma unroll
            for (int j = 0; j < 8; ++j) {
                int r = rt * 16 + 8 * lg + j;
                if (r < rows) {
                    float v = acc[rt][j];
                    float g = 0.5f * v * (1.0f + erff(v * 0.7071067811865475f));
                    hout[(size_t)r * H_DIM + ncol] = (_Float16)g;
                }
            }
    }
}

// -------------------------------------------------------------------
// fc2: o = h @ W2^T + b2, softmax over D, combined += gate*softmax.
// One (expert, 32-token tile) per block, 512 threads = 16 waves.
__global__ __launch_bounds__(512)
void moe_fc2_kernel(const _Float16* __restrict__ hbuf,
                    const _Float16* __restrict__ w2h,
                    const float* __restrict__ b2,
                    const float* __restrict__ gateval,
                    const int* __restrict__ perm,
                    const int* __restrict__ countI,
                    const int* __restrict__ baseI,
                    float* __restrict__ outc) {
    __shared__ float pmax[16 * 32];
    __shared__ float psum[16 * 32];
    __shared__ float rowmax[32];
    __shared__ float rowsum[32];

    const int e = blockIdx.y;
    const int t = blockIdx.x;
    const int cnt = countI[e];
    if (t * 32 >= cnt) return;
    const int rows = (cnt - t * 32 < 32) ? (cnt - t * 32) : 32;
    const int base = baseI[e];

    const int tid  = threadIdx.x;
    const int lane = tid & 31;
    const int wv   = tid >> 5;
    const int lg   = lane >> 4;
    const int ln   = lane & 15;

    const _Float16* hrow0 = hbuf + (size_t)(base + t * 32) * H_DIM;
    const _Float16* w2e   = w2h + (size_t)e * D_DIM * H_DIM;
    const int dbase = wv * 64;

    v8f acc[2][4];
#pragma unroll
    for (int ct = 0; ct < 4; ++ct) {
        float bv = b2[e * D_DIM + dbase + ct * 16 + ln];
#pragma unroll
        for (int j = 0; j < 8; ++j) { acc[0][ct][j] = bv; acc[1][ct][j] = bv; }
    }

    int r0 = ln;      if (r0 >= rows) r0 = rows - 1;
    int r1 = 16 + ln; if (r1 >= rows) r1 = rows - 1;
    const _Float16* ar0 = hrow0 + (size_t)r0 * H_DIM + lg * 16;
    const _Float16* ar1 = hrow0 + (size_t)r1 * H_DIM + lg * 16;

    for (int kt = 0; kt < H_DIM / 32; ++kt) {
        const int kb = kt * 32;
        union { v16h v; v8h h[2]; } A0, A1;
        A0.h[0] = *(const v8h*)(ar0 + kb);
        A0.h[1] = *(const v8h*)(ar0 + kb + 8);
        A1.h[0] = *(const v8h*)(ar1 + kb);
        A1.h[1] = *(const v8h*)(ar1 + kb + 8);
#pragma unroll
        for (int ct = 0; ct < 4; ++ct) {
            const int ncol = dbase + ct * 16 + ln;
            v16h Bv = *(const v16h*)(w2e + (size_t)ncol * H_DIM + lg * 16 + kb);
            acc[0][ct] = __builtin_amdgcn_wmma_f32_16x16x32_f16(false, A0.v, false, Bv,
                                                                (short)0, acc[0][ct], false, false);
            acc[1][ct] = __builtin_amdgcn_wmma_f32_16x16x32_f16(false, A1.v, false, Bv,
                                                                (short)0, acc[1][ct], false, false);
        }
    }

    // ---- cross-wave softmax: row max ----
    float m[2][8];
#pragma unroll
    for (int rt = 0; rt < 2; ++rt)
#pragma unroll
        for (int j = 0; j < 8; ++j) {
            float v = acc[rt][0][j];
            v = fmaxf(v, acc[rt][1][j]); v = fmaxf(v, acc[rt][2][j]); v = fmaxf(v, acc[rt][3][j]);
            m[rt][j] = v;
        }
#pragma unroll
    for (int off = 1; off < 16; off <<= 1)
#pragma unroll
        for (int rt = 0; rt < 2; ++rt)
#pragma unroll
            for (int j = 0; j < 8; ++j)
                m[rt][j] = fmaxf(m[rt][j], __shfl_xor(m[rt][j], off, 32));
    if (ln == 0) {
#pragma unroll
        for (int rt = 0; rt < 2; ++rt)
#pragma unroll
            for (int j = 0; j < 8; ++j)
                pmax[wv * 32 + rt * 16 + 8 * lg + j] = m[rt][j];
    }
    __syncthreads();
    if (tid < 32) {
        float v = -3.4e38f;
        for (int w = 0; w < 16; ++w) v = fmaxf(v, pmax[w * 32 + tid]);
        rowmax[tid] = v;
    }
    __syncthreads();

    // ---- exp and row sum ----
    float s[2][8];
#pragma unroll
    for (int rt = 0; rt < 2; ++rt)
#pragma unroll
        for (int j = 0; j < 8; ++j) {
            const float rm = rowmax[rt * 16 + 8 * lg + j];
            float sum = 0.0f;
#pragma unroll
            for (int ct = 0; ct < 4; ++ct) {
                float ev = __expf(acc[rt][ct][j] - rm);
                acc[rt][ct][j] = ev;
                sum += ev;
            }
            s[rt][j] = sum;
        }
#pragma unroll
    for (int off = 1; off < 16; off <<= 1)
#pragma unroll
        for (int rt = 0; rt < 2; ++rt)
#pragma unroll
            for (int j = 0; j < 8; ++j)
                s[rt][j] += __shfl_xor(s[rt][j], off, 32);
    if (ln == 0) {
#pragma unroll
        for (int rt = 0; rt < 2; ++rt)
#pragma unroll
            for (int j = 0; j < 8; ++j)
                psum[wv * 32 + rt * 16 + 8 * lg + j] = s[rt][j];
    }
    __syncthreads();
    if (tid < 32) {
        float v = 0.0f;
        for (int w = 0; w < 16; ++w) v += psum[w * 32 + tid];
        rowsum[tid] = v;
    }
    __syncthreads();

    // ---- combined[b,:] += gate * softmax(o) ----
#pragma unroll
    for (int rt = 0; rt < 2; ++rt)
#pragma unroll
        for (int j = 0; j < 8; ++j) {
            const int r = rt * 16 + 8 * lg + j;
            if (r < rows) {
                const int p = perm[base + t * 32 + r];
                const int tok = p >> 1;
                const float f = gateval[p] / rowsum[r];
#pragma unroll
                for (int ct = 0; ct < 4; ++ct) {
                    const int col = dbase + ct * 16 + ln;
                    unsafeAtomicAdd(&outc[(size_t)tok * D_DIM + col], acc[rt][ct][j] * f);
                }
            }
        }
}

// -------------------------------------------------------------------
__global__ __launch_bounds__(256)
void log_kernel(float* out, int n) {
    int i = blockIdx.x * blockDim.x + threadIdx.x;
    int stride = gridDim.x * blockDim.x;
    for (int j = i; j < n; j += stride) {
        float c = out[j];
        c = (c == 0.0f) ? EPS_COMBINE : c;
        out[j] = __logf(c);
    }
}

// -------------------------------------------------------------------
extern "C" void kernel_launch(void* const* d_in, const int* in_sizes, int n_in,
                              void* d_out, int out_size, void* d_ws, size_t ws_size,
                              hipStream_t stream) {
    const float* x  = (const float*)d_in[0];
    const float* wg = (const float*)d_in[1];
    const float* w1 = (const float*)d_in[2];
    const float* b1 = (const float*)d_in[3];
    const float* w2 = (const float*)d_in[4];
    const float* b2 = (const float*)d_in[5];
    float* out = (float*)d_out;   // [B*D] combined/y, then [1] loss

    char* ws = (char*)d_ws;
    _Float16* w1h  = (_Float16*)ws;
    _Float16* w2h  = (_Float16*)(ws + WS_W2H);
    _Float16* x16  = (_Float16*)(ws + WS_X16);
    _Float16* hbuf = (_Float16*)(ws + WS_HBUF);
    float* gateval = (float*)(ws + WS_GVAL);
    int*   gidx    = (int*)(ws + WS_GIDX);
    int*   perm    = (int*)(ws + WS_PERM);
    float* importance = (float*)(ws + WS_CTR);
    int*   countI  = (int*)(ws + WS_CTR + 32);
    int*   cursor  = (int*)(ws + WS_CTR + 64);
    int*   baseI   = (int*)(ws + WS_CTR + 96);

    zero_kernel<<<2048, 256, 0, stream>>>(out, B_TOK * D_DIM, (int*)(ws + WS_CTR), 24);
    cvt_f32_to_f16_kernel<<<2048, 256, 0, stream>>>(w1, w1h, E_EXP * H_DIM * D_DIM);
    cvt_f32_to_f16_kernel<<<2048, 256, 0, stream>>>(w2, w2h, E_EXP * D_DIM * H_DIM);
    cvt_f32_to_f16_kernel<<<2048, 256, 0, stream>>>(x, x16, B_TOK * D_DIM);
    gate_kernel<<<B_TOK / 8, 256, 0, stream>>>(x, wg, gateval, gidx, importance, countI);
    finalize_stats_kernel<<<1, 1, 0, stream>>>(importance, countI, baseI,
                                               out + (size_t)B_TOK * D_DIM);
    scatter_kernel<<<B_TOK / 256, 256, 0, stream>>>(gidx, baseI, cursor, perm);

    dim3 grid1(256, E_EXP);    // 64-token tiles, upper bound; early exit past count
    moe_fc1_kernel<<<grid1, 512, 131072, stream>>>(x16, w1h, b1, perm, countI, baseI, hbuf);

    dim3 grid2(512, E_EXP);    // 32-token tiles, upper bound; early exit past count
    moe_fc2_kernel<<<grid2, 512, 0, stream>>>(hbuf, w2h, b2, gateval, perm, countI, baseI, out);

    log_kernel<<<4096, 256, 0, stream>>>(out, B_TOK * D_DIM);
}
